// rnn_lyapunov_70308614636158
// MI455X (gfx1250) — matrix operand order repaired and checked by reading, not compiled, not run on hardware
//
#include <hip/hip_runtime.h>
#include <hip/hip_bf16.h>
#include <stdint.h>

// ---------------------------------------------------------------------------
// Elman RNN w/ equilibrium init, CDNA5 (gfx1250) bf16 WMMA implementation.
//   B=64, T=2048, NIN=64, NH=512, NOUT=64, FP_ITERS=60
//
// Pipeline:
//   k1: weight fusion/convert (A = W_hh*diag(omega^2) -> bf16, etc.) [parallel]
//   k2: u -> bf16                                                    [parallel]
//   k3: recurrence, 4 WGs (16 batch rows each).  h (16x512 bf16) and
//       9/16 of A (k-tiles 7..15, bank-conflict-padded) fill the WGP's
//       320 KB LDS; 7 k-tiles stream from L2 in a throttled (unroll 2)
//       loop to keep register demand ~250 (no scratch spills).        [serial]
//   k4: output head hs @ W_lin^T + b_lin, full-chip WMMA              [parallel]
// ---------------------------------------------------------------------------

#define RNN_B    64
#define RNN_T    2048
#define RNN_NIN  64
#define RNN_NH   512
#define RNN_NOUT 64
#define RNN_FPIT 60

#define KT_TOT   (RNN_NH / 32)   // 16 k-tiles
#define KT_G     7               // k-tiles 0..6 stream from L2
#define KT_L     (KT_TOT - KT_G) // k-tiles 7..15 (9) staged in LDS
#define A_COLS   (KT_L * 32)     // 288 elements per LDS row
#define A_ROW    (A_COLS + 8)    // padded stride: 296 elem = 592 B = 148 dw
                                 // -> 20-bank rotation/row, conflict-free

typedef __bf16 bf16_t;
typedef __attribute__((ext_vector_type(16))) __bf16 v16bf;
typedef __attribute__((ext_vector_type(8)))  float  v8f;

struct Pair128 { uint4 lo, hi; };

static __device__ __forceinline__ v16bf make_frag(uint4 lo, uint4 hi) {
    Pair128 p{lo, hi};
    return __builtin_bit_cast(v16bf, p);
}

// A-operand fragment (16x32 bf16): per lane M = lane&15; 16 contiguous K
// starting at kt*32 + (lane<16 ? 0 : 16), given k0 = kt*32 + (lane<16 ? 0 : 8).
static __device__ __forceinline__ v16bf load_afrag(const bf16_t* rowBase, int k0) {
    uint4 lo = *(const uint4*)(rowBase + k0 + (k0 & 8));
    uint4 hi = *(const uint4*)(rowBase + k0 + (k0 & 8) + 8);
    return make_frag(lo, hi);
}

// B-operand fragment (32x16 bf16): per lane N fixed, 16 contiguous K values
// starting at k0 = kt*32 + (lane<16 ? 0 : 16).  p = &M[n*K + k0].
static __device__ __forceinline__ v16bf load_bfrag(const bf16_t* p) {
    uint4 lo = *(const uint4*)(p);
    uint4 hi = *(const uint4*)(p + 8);
    return make_frag(lo, hi);
}

static __device__ __forceinline__ v8f wmma_bf16(v16bf a, v16bf b, v8f c) {
    return __builtin_amdgcn_wmma_f32_16x16x32_bf16(
        /*neg_a=*/false, a, /*neg_b=*/false, b,
        /*c_mod=*/(short)0, c, /*reuse_a=*/false, /*reuse_b=*/false);
}

static __device__ __forceinline__ v8f splat8(float x) {
    v8f r;
#pragma unroll
    for (int i = 0; i < 8; ++i) r[i] = x;
    return r;
}

static __device__ __forceinline__ float fast_tanh(float x) {
    // tanh(x) = 1 - 2/(exp(2x)+1); exp -> v_exp_f32 (handles +-inf limits)
    float e = __expf(2.0f * x);
    return 1.0f - 2.0f / (e + 1.0f);
}

// ---------------------------------------------------------------------------
// k1: weight conversion / fusion
// ---------------------------------------------------------------------------
__global__ __launch_bounds__(256) void rnn_prep_weights(
    const float* __restrict__ Whh, const float* __restrict__ omega,
    const float* __restrict__ Wih, const float* __restrict__ Wlin,
    const float* __restrict__ bih, const float* __restrict__ bhh,
    bf16_t* __restrict__ A_bf, bf16_t* __restrict__ Wih_bf,
    bf16_t* __restrict__ Wlin_bf, float* __restrict__ b_ws)
{
    int i = blockIdx.x * 256 + threadIdx.x;
    if (i < RNN_NH * RNN_NH) {
        int k = i & (RNN_NH - 1);
        float om = omega[k];
        A_bf[i] = (bf16_t)(Whh[i] * om * om);   // A[n][k] = Whh[n][k]*omega[k]^2
    }
    if (i < RNN_NH * RNN_NIN) {
        Wih_bf[i] = (bf16_t)Wih[i];             // [NH][NIN] row-major
    }
    if (i < RNN_NOUT * RNN_NH) {
        Wlin_bf[i] = (bf16_t)Wlin[i];           // [NOUT][NH] row-major
    }
    if (i < RNN_NH) {
        b_ws[i] = bih[i] + bhh[i];
    }
}

// ---------------------------------------------------------------------------
// k2: input conversion u f32 -> bf16
// ---------------------------------------------------------------------------
__global__ __launch_bounds__(256) void rnn_prep_u(
    const float* __restrict__ u, bf16_t* __restrict__ u_bf, int n)
{
    int i = blockIdx.x * 256 + threadIdx.x;
    if (i < n) u_bf[i] = (bf16_t)u[i];
}

// ---------------------------------------------------------------------------
// k3: fixed-point init + sequential recurrence.
//   4 workgroups, each owns batch rows [16*bg, 16*bg+16).
//   8 waves/WG; wave w owns N-tiles {4w..4w+3} (N = 512 = 32 tiles of 16).
//   LDS: h (16x512 bf16) + A k-tiles 7..15 (512 rows x 296 padded).
//   A k-tiles 0..6 + all W_ih fragments stream from L2.
// ---------------------------------------------------------------------------
__global__ __launch_bounds__(256) void rnn_recurrence(
    const bf16_t* __restrict__ u_bf,   // [B, T, NIN]
    const bf16_t* __restrict__ A_bf,   // [NH, NH] row-major [n][k]
    const bf16_t* __restrict__ Wih_bf, // [NH, NIN]
    const float*  __restrict__ bvec,   // [NH]
    bf16_t*       __restrict__ hs)     // [B, T, NH]
{
    __shared__ __align__(16) bf16_t h_lds[16][RNN_NH];       //  16 KB
    __shared__ __align__(16) bf16_t a_lds[RNN_NH][A_ROW];    // 296 KB

    const int bg   = blockIdx.x;          // batch group 0..3
    const int tid  = threadIdx.x;
    const int wave = tid >> 5;            // 0..7
    const int lane = tid & 31;
    const int l16  = lane & 15;
    const int aOff = (lane < 16) ? 0 : 8;   // A-frag K offset / D-frag M offset
    const int bOff = (lane < 16) ? 0 : 16;  // B-frag K offset

    // bias splat values per owned N-tile (C/D layout: N = lane&15)
    float bv[4];
#pragma unroll
    for (int j = 0; j < 4; ++j) bv[j] = bvec[(wave * 4 + j) * 16 + l16];

    // per-N-tile weight row bases (per-lane N)
    const bf16_t* aRowG[4];   // streamed base: &A[n][bOff]
    const bf16_t* aRowL[4];   // LDS base:      &a_lds[n][bOff]
    const bf16_t* iRowG[4];   // W_ih base:     &W_ih[n][bOff]
#pragma unroll
    for (int j = 0; j < 4; ++j) {
        const int n = (wave * 4 + j) * 16 + l16;
        aRowG[j] = A_bf + (size_t)n * RNN_NH + bOff;
        aRowL[j] = &a_lds[n][bOff];
        iRowG[j] = Wih_bf + (size_t)n * RNN_NIN + bOff;
    }

    // zero h
    for (int i = tid; i < (16 * RNN_NH) / 8; i += 256)
        ((uint4*)h_lds)[i] = uint4{0u, 0u, 0u, 0u};

    // stage A k-tiles 7..15 into LDS: a_lds[n][c] = A[n][KT_G*32 + c],
    // c in [0, 288); copied as 8-element (16 B) chunks, 36 per row.
    for (int idx = tid; idx < RNN_NH * (A_COLS / 8); idx += 256) {
        int n = idx / (A_COLS / 8);
        int c = idx - n * (A_COLS / 8);
        *(uint4*)(&a_lds[n][c * 8]) =
            *(const uint4*)(A_bf + (size_t)n * RNN_NH + KT_G * 32 + c * 8);
    }
    __syncthreads();

    // ---- c0 = b + u[:,0,:] @ W_ih^T  (kept in registers per wave) ----
    v8f c0[4];
#pragma unroll
    for (int j = 0; j < 4; ++j) c0[j] = splat8(bv[j]);
    {
        const bf16_t* uRow = u_bf + ((size_t)(bg * 16 + l16) * RNN_T + 0) * RNN_NIN;
#pragma unroll
        for (int kt = 0; kt < RNN_NIN / 32; ++kt) {
            v16bf af = load_afrag(uRow, kt * 32 + aOff);
#pragma unroll
            for (int j = 0; j < 4; ++j)
                c0[j] = wmma_bf16(af, load_bfrag(iRowG[j] + kt * 32), c0[j]);
        }
    }

    const bf16_t* hRow = &h_lds[l16][0];

    // ---- equilibrium fixed point: x = tanh(x @ A^T + c0), 60 iters ----
    for (int it = 0; it < RNN_FPIT; ++it) {
        v8f acc[4];
#pragma unroll
        for (int j = 0; j < 4; ++j) acc[j] = c0[j];
        // k-tiles 0..6: B-frags streamed from L2 (throttled prefetch)
#pragma unroll 2
        for (int kt = 0; kt < KT_G; ++kt) {
            v16bf af = load_afrag(hRow, kt * 32 + aOff);
#pragma unroll
            for (int j = 0; j < 4; ++j)
                acc[j] = wmma_bf16(af, load_bfrag(aRowG[j] + kt * 32), acc[j]);
        }
        // k-tiles 7..15: B-frags from LDS
#pragma unroll
        for (int kt = 0; kt < KT_L; ++kt) {
            v16bf af = load_afrag(hRow, (KT_G + kt) * 32 + aOff);
#pragma unroll
            for (int j = 0; j < 4; ++j)
                acc[j] = wmma_bf16(af, load_bfrag(aRowL[j] + kt * 32), acc[j]);
        }
        __syncthreads();   // all reads of h done before overwrite
#pragma unroll
        for (int j = 0; j < 4; ++j) {
            int n = (wave * 4 + j) * 16 + l16;
#pragma unroll
            for (int v = 0; v < 8; ++v)
                h_lds[aOff + v][n] = (bf16_t)fast_tanh(acc[j][v]);
        }
        __syncthreads();   // h fully updated before next iteration reads
    }

    // ---- main recurrence over time ----
    for (int t = 0; t < RNN_T; ++t) {
        v8f acc[4];
#pragma unroll
        for (int j = 0; j < 4; ++j) acc[j] = splat8(bv[j]);

        // fused input projection: + u_t @ W_ih^T
        const bf16_t* uRow = u_bf + ((size_t)(bg * 16 + l16) * RNN_T + t) * RNN_NIN;
#pragma unroll
        for (int kt = 0; kt < RNN_NIN / 32; ++kt) {
            v16bf af = load_afrag(uRow, kt * 32 + aOff);
#pragma unroll
            for (int j = 0; j < 4; ++j)
                acc[j] = wmma_bf16(af, load_bfrag(iRowG[j] + kt * 32), acc[j]);
        }
        // + h_{t-1} @ A^T : k-tiles 0..6 streamed from L2
#pragma unroll 2
        for (int kt = 0; kt < KT_G; ++kt) {
            v16bf af = load_afrag(hRow, kt * 32 + aOff);
#pragma unroll
            for (int j = 0; j < 4; ++j)
                acc[j] = wmma_bf16(af, load_bfrag(aRowG[j] + kt * 32), acc[j]);
        }
        // + h_{t-1} @ A^T : k-tiles 7..15 from LDS
#pragma unroll
        for (int kt = 0; kt < KT_L; ++kt) {
            v16bf af = load_afrag(hRow, (KT_G + kt) * 32 + aOff);
#pragma unroll
            for (int j = 0; j < 4; ++j)
                acc[j] = wmma_bf16(af, load_bfrag(aRowL[j] + kt * 32), acc[j]);
        }
        __syncthreads();
#pragma unroll
        for (int j = 0; j < 4; ++j) {
            int n = (wave * 4 + j) * 16 + l16;
#pragma unroll
            for (int v = 0; v < 8; ++v)
                h_lds[aOff + v][n] = (bf16_t)fast_tanh(acc[j][v]);
        }
        __syncthreads();

        // stream h_t (coalesced 16B chunks) to hs[b, t, :]
        const uint4* src = (const uint4*)h_lds;
        for (int i = tid; i < (16 * RNN_NH) / 8; i += 256) {
            int m = i >> 6;          // row 0..15  (64 uint4 per 512-bf16 row)
            int c = i & 63;
            uint4* dst = (uint4*)(hs + ((size_t)(bg * 16 + m) * RNN_T + t) * RNN_NH) + c;
            *dst = src[i];
        }
        // next iteration's LDS writes are fenced by its own first barrier
    }
}

// ---------------------------------------------------------------------------
// k4: output head  out[b,t,:] = hs[b,t,:] @ W_lin^T + b_lin
//   one wave per 16-row M-tile (M = B*T = 131072 rows -> 8192 waves)
// ---------------------------------------------------------------------------
__global__ __launch_bounds__(256) void rnn_head(
    const bf16_t* __restrict__ hs,      // [B*T, NH]
    const bf16_t* __restrict__ Wlin_bf, // [NOUT, NH]
    const float*  __restrict__ blin,    // [NOUT]
    float*        __restrict__ out)     // [B*T, NOUT]
{
    const int wid  = (blockIdx.x * 256 + threadIdx.x) >> 5;  // M-tile id
    const int lane = threadIdx.x & 31;
    const int l16  = lane & 15;
    const int aOff = (lane < 16) ? 0 : 8;
    const int bOff = (lane < 16) ? 0 : 16;

    v8f acc[4];
#pragma unroll
    for (int j = 0; j < 4; ++j) acc[j] = splat8(blin[j * 16 + l16]);

    const bf16_t* aRow = hs + ((size_t)wid * 16 + l16) * RNN_NH;
#pragma unroll 4
    for (int kt = 0; kt < RNN_NH / 32; ++kt) {
        v16bf af = load_afrag(aRow, kt * 32 + aOff);
#pragma unroll
        for (int j = 0; j < 4; ++j) {
            const bf16_t* bp = Wlin_bf +
                (size_t)(j * 16 + l16) * RNN_NH + kt * 32 + bOff;
            acc[j] = wmma_bf16(af, load_bfrag(bp), acc[j]);
        }
    }

    const size_t rBase = (size_t)wid * 16;
#pragma unroll
    for (int j = 0; j < 4; ++j) {
        int n = j * 16 + l16;
#pragma unroll
        for (int v = 0; v < 8; ++v)
            out[(rBase + aOff + v) * RNN_NOUT + n] = acc[j][v];
    }
}

// ---------------------------------------------------------------------------
// host launcher
// ---------------------------------------------------------------------------
extern "C" void kernel_launch(void* const* d_in, const int* in_sizes, int n_in,
                              void* d_out, int out_size, void* d_ws, size_t ws_size,
                              hipStream_t stream) {
    const float* u     = (const float*)d_in[0];
    const float* W_ih  = (const float*)d_in[1];
    const float* W_hh  = (const float*)d_in[2];
    const float* b_ih  = (const float*)d_in[3];
    const float* b_hh  = (const float*)d_in[4];
    const float* omega = (const float*)d_in[5];
    const float* W_lin = (const float*)d_in[6];
    const float* b_lin = (const float*)d_in[7];
    float* out = (float*)d_out;

    // workspace carve-up (all offsets 256B aligned)
    char* ws = (char*)d_ws;
    size_t off = 0;
    bf16_t* A_bf    = (bf16_t*)(ws + off); off += (size_t)RNN_NH * RNN_NH * 2;      // 512 KB
    bf16_t* Wih_bf  = (bf16_t*)(ws + off); off += (size_t)RNN_NH * RNN_NIN * 2;     // 64 KB
    bf16_t* Wlin_bf = (bf16_t*)(ws + off); off += (size_t)RNN_NOUT * RNN_NH * 2;    // 64 KB
    float*  b_ws    = (float*) (ws + off); off += (size_t)RNN_NH * 4;               // 2 KB
    off = (off + 255) & ~(size_t)255;
    bf16_t* u_bf    = (bf16_t*)(ws + off); off += (size_t)RNN_B * RNN_T * RNN_NIN * 2;  // 16 MB
    bf16_t* hs      = (bf16_t*)(ws + off); off += (size_t)RNN_B * RNN_T * RNN_NH * 2;   // 128 MB
    if (ws_size < off) return;  // workspace too small: nothing safe to do

    // k1: weights
    rnn_prep_weights<<<(RNN_NH * RNN_NH + 255) / 256, 256, 0, stream>>>(
        W_hh, omega, W_ih, W_lin, b_ih, b_hh, A_bf, Wih_bf, Wlin_bf, b_ws);

    // k2: inputs
    const int nU = RNN_B * RNN_T * RNN_NIN;
    rnn_prep_u<<<(nU + 255) / 256, 256, 0, stream>>>(u, u_bf, nU);

    // k3: sequential recurrence (4 WGs, one per 16-row batch group)
    rnn_recurrence<<<RNN_B / 16, 256, 0, stream>>>(u_bf, A_bf, Wih_bf, b_ws, hs);

    // k4: output head (8192 M-tiles, 8 waves/block)
    const int headWaves = (RNN_B * RNN_T) / 16;          // 8192
    rnn_head<<<headWaves / 8, 256, 0, stream>>>(hs, Wlin_bf, b_lin, out);
}
